// Discriminator_1039382086148
// MI455X (gfx1250) — matrix operand order repaired and checked by reading, not compiled, not run on hardware
//
#include <hip/hip_runtime.h>

#define NNODES 100000
#define DIM    128

typedef __attribute__((ext_vector_type(16))) __bf16 bf16x16;
typedef __attribute__((ext_vector_type(8)))  float  floatx8;

__device__ __forceinline__ unsigned short f32_to_bf16_rne(float f) {
  union { float f; unsigned u; } v; v.f = f;
  unsigned r = v.u + 0x7FFFu + ((v.u >> 16) & 1u);   // round-to-nearest-even
  return (unsigned short)(r >> 16);
}

// ---------------------------------------------------------------------------
// H[M x 128] = X[M x 128] @ W[128 x 128]   (bf16 WMMA, f32 accumulate)
// One block = 16 rows x 128 cols; 8 wave32s, each owns a 16x16 tile column.
// ---------------------------------------------------------------------------
__global__ __launch_bounds__(256) void gemm_wmma(
    const float* __restrict__ X, const float* __restrict__ W,
    float* __restrict__ H) {
  __shared__ __align__(16) unsigned short sX[16][136];   // X tile, bf16
  __shared__ __align__(16) unsigned short sW[128][136];  // W transposed: sW[n][k]

  const int tid  = threadIdx.x;
  const int row0 = blockIdx.x * 16;

  // Stage W (transposed) as bf16: coalesced global read, strided LDS write.
  for (int idx = tid; idx < 128 * 128; idx += 256) {
    int k = idx >> 7, n = idx & 127;
    sW[n][k] = f32_to_bf16_rne(W[idx]);
  }
  // Stage 16x128 X tile as bf16.
  for (int idx = tid; idx < 16 * 128; idx += 256) {
    int m = idx >> 7, k = idx & 127;
    sX[m][k] = f32_to_bf16_rne(X[(size_t)(row0 + m) * DIM + k]);
  }
  __syncthreads();

  const int wave = tid >> 5;          // wave32: 8 waves per block
  const int lane = tid & 31;
  const int n0   = wave * 16;

  union Frag { bf16x16 v; uint4 q[2]; };

  const int mA    = lane & 15;        // A: lane = M row
  const int kAoff = (lane >> 4) * 8;  // halves 0-7: K=kA..+7 ; 8-15: K+16
  const int nB    = n0 + (lane & 15); // B: lane = N col
  const int kBoff = (lane >> 4) * 8;  // halves 0-7: K=kB..+7 ; 8-15: K+16

  floatx8 c = {};
  #pragma unroll
  for (int k0 = 0; k0 < 128; k0 += 32) {
    Frag a, b;
    a.q[0] = *(const uint4*)&sX[mA][k0 + kAoff];
    a.q[1] = *(const uint4*)&sX[mA][k0 + 16 + kAoff];
    b.q[0] = *(const uint4*)&sW[nB][k0 + kBoff];
    b.q[1] = *(const uint4*)&sW[nB][k0 + 16 + kBoff];
    c = __builtin_amdgcn_wmma_f32_16x16x32_bf16(
        false, a.v, false, b.v, (short)0, c, false, false);
  }

  // D layout: lanes 0-15 -> M = r ; lanes 16-31 -> M = 8 + r ; N = lane&15
  const int nn    = n0 + (lane & 15);
  const int mbase = (lane >> 4) * 8;
  #pragma unroll
  for (int r = 0; r < 8; ++r)
    H[(size_t)(row0 + mbase + r) * DIM + nn] = c[r];
}

// ---------------------------------------------------------------------------
// Degree / normalization
// ---------------------------------------------------------------------------
__global__ void deg_count(const int* __restrict__ dst, float* __restrict__ deg, int E) {
  int i = blockIdx.x * blockDim.x + threadIdx.x;
  if (i < E) atomicAdd(&deg[dst[i]], 1.0f);
}

__global__ void deg_finalize(float* __restrict__ deg, int n) {
  int i = blockIdx.x * blockDim.x + threadIdx.x;
  if (i < n) deg[i] = rsqrtf(deg[i] + 1.0f);   // +1 self-loop; always > 0
}

// ---------------------------------------------------------------------------
// One wave32 per edge: lane l handles features [4l, 4l+3].
// acc[dst] += dinv[src]*dinv[dst] * h[src]
// ---------------------------------------------------------------------------
__global__ __launch_bounds__(256) void edge_scatter(
    const float* __restrict__ h, const float* __restrict__ dinv,
    const int* __restrict__ src, const int* __restrict__ dst,
    float* __restrict__ acc, int E) {
  int gtid = blockIdx.x * blockDim.x + threadIdx.x;
  int e    = gtid >> 5;
  int lane = gtid & 31;
  if (e >= E) return;
  int s = src[e], d = dst[e];
  float norm = dinv[s] * dinv[d];
  float4 v = ((const float4*)(h + (size_t)s * DIM))[lane];
  float* o = acc + (size_t)d * DIM + lane * 4;
  atomicAdd(o + 0, v.x * norm);
  atomicAdd(o + 1, v.y * norm);
  atomicAdd(o + 2, v.z * norm);
  atomicAdd(o + 3, v.w * norm);
}

// acc = relu(acc + dinv^2 * h + bias)   (self-loop term + bias + activation)
__global__ void node_finalize(float* __restrict__ acc, const float* __restrict__ h,
                              const float* __restrict__ dinv,
                              const float* __restrict__ bias, int total) {
  int idx = blockIdx.x * blockDim.x + threadIdx.x;
  if (idx >= total) return;
  int node = idx >> 7, col = idx & 127;
  float di = dinv[node];
  float v  = acc[idx] + di * di * h[idx] + bias[col];
  acc[idx] = v > 0.0f ? v : 0.0f;
}

// out[node] = dot(act[node, :], Wf) + bf   — one wave32 per node
__global__ __launch_bounds__(256) void final_dot(
    const float* __restrict__ act, const float* __restrict__ Wf,
    const float* __restrict__ bf, float* __restrict__ out, int n) {
  int gtid = blockIdx.x * blockDim.x + threadIdx.x;
  int node = gtid >> 5;
  int lane = gtid & 31;
  if (node >= n) return;
  float4 a = ((const float4*)(act + (size_t)node * DIM))[lane];
  float4 w = ((const float4*)Wf)[lane];
  float p = a.x * w.x + a.y * w.y + a.z * w.z + a.w * w.w;
  #pragma unroll
  for (int o = 16; o; o >>= 1) p += __shfl_xor(p, o, 32);
  if (lane == 0) out[node] = p + bf[0];
}

// ---------------------------------------------------------------------------
extern "C" void kernel_launch(void* const* d_in, const int* in_sizes, int n_in,
                              void* d_out, int out_size, void* d_ws, size_t ws_size,
                              hipStream_t stream) {
  const float* x  = (const float*)d_in[0];
  const float* W1 = (const float*)d_in[1];
  const float* b1 = (const float*)d_in[2];
  const float* W2 = (const float*)d_in[3];
  const float* b2 = (const float*)d_in[4];
  const float* Wf = (const float*)d_in[5];
  const float* bf = (const float*)d_in[6];
  const int*   ei = (const int*)d_in[7];

  const int E = in_sizes[7] / 2;
  const int* src = ei;
  const int* dst = ei + E;
  float* out = (float*)d_out;

  char*  ws   = (char*)d_ws;
  float* dinv = (float*)ws;
  size_t off  = ((size_t)NNODES * sizeof(float) + 255) & ~(size_t)255;
  float* bufA = (float*)(ws + off);                 // GEMM output h
  float* bufB = bufA + (size_t)NNODES * DIM;        // aggregation / activation
  const size_t featBytes = (size_t)NNODES * DIM * sizeof(float);

  // deg^-1/2 (self-loops included)
  hipMemsetAsync(dinv, 0, (size_t)NNODES * sizeof(float), stream);
  deg_count<<<(E + 255) / 256, 256, 0, stream>>>(dst, dinv, E);
  deg_finalize<<<(NNODES + 255) / 256, 256, 0, stream>>>(dinv, NNODES);

  // ---- Layer 1: h1 = x @ W1 ; agg ; relu ----
  gemm_wmma<<<NNODES / 16, 256, 0, stream>>>(x, W1, bufA);
  hipMemsetAsync(bufB, 0, featBytes, stream);
  edge_scatter<<<(E + 7) / 8, 256, 0, stream>>>(bufA, dinv, src, dst, bufB, E);
  node_finalize<<<(NNODES * DIM + 255) / 256, 256, 0, stream>>>(bufB, bufA, dinv, b1,
                                                                NNODES * DIM);

  // ---- Layer 2: h2 = act1 @ W2 ; agg ; relu ----
  gemm_wmma<<<NNODES / 16, 256, 0, stream>>>(bufB, W2, bufA);
  hipMemsetAsync(bufB, 0, featBytes, stream);   // act1 consumed; reuse buffer
  edge_scatter<<<(E + 7) / 8, 256, 0, stream>>>(bufA, dinv, src, dst, bufB, E);
  node_finalize<<<(NNODES * DIM + 255) / 256, 256, 0, stream>>>(bufB, bufA, dinv, b2,
                                                                NNODES * DIM);

  // ---- Readout: out = act2 @ Wf + bf ----
  final_dot<<<(NNODES * 32 + 255) / 256, 256, 0, stream>>>(bufB, Wf, bf, out, NNODES);
}
